// MultiScaleRetention_51702816309254
// MI455X (gfx1250) — matrix operand order repaired and checked by reading, not compiled, not run on hardware
//
#include <hip/hip_runtime.h>

// ---------------- problem constants ----------------
#define Bsz 2
#define Tlen 2048
#define HS 1024
#define QK 1024
#define VD 2048
#define NH 8
#define HD_QK 128
#define HD_V 256
// ---------------------------------------------------

typedef unsigned short u16;
typedef __bf16 bf16_t;
typedef __bf16 v16bf __attribute__((ext_vector_type(16)));
typedef float  v8f   __attribute__((ext_vector_type(8)));
typedef unsigned int v4u __attribute__((ext_vector_type(4)));

union Frag { v4u q[2]; v16bf v; };   // 32B: 16 bf16 = one WMMA A/B operand

__device__ __forceinline__ u16 f2bf(float f) {
    unsigned u = __float_as_uint(f);
    unsigned r = u + 0x7fffu + ((u >> 16) & 1u);   // round-to-nearest-even
    return (u16)(r >> 16);
}
__device__ __forceinline__ float bf2f(u16 u) {
    return __uint_as_float(((unsigned)u) << 16);
}

// =====================================================================
// f32 -> bf16 pack (grid-stride)
// =====================================================================
__global__ void f32_to_bf16_kernel(const float* __restrict__ in, u16* __restrict__ out, int n) {
    for (int i = blockIdx.x * blockDim.x + threadIdx.x; i < n; i += gridDim.x * blockDim.x)
        out[i] = f2bf(in[i]);
}

// =====================================================================
// Generic bf16 GEMM: C[M,N] = A[M,K] * B[N,K]^T  (both row-major, K contig)
// block tile 128x128, 256 thr = 8 waves (2 M x 4 N), wave tile 64x32
// =====================================================================
__global__ __launch_bounds__(256) void gemm_bf16_kernel(
    const u16* __restrict__ A, const u16* __restrict__ Bm,
    float* __restrict__ C, int M, int N, int K)
{
    __shared__ __align__(16) u16 As[128][40];
    __shared__ __align__(16) u16 Bs[128][40];

    const int m0 = blockIdx.y * 128, n0 = blockIdx.x * 128;
    const int tid  = threadIdx.x;
    const int wave = tid >> 5, lane = tid & 31;
    const int wm = (wave >> 2) * 64;   // 0 or 64
    const int wn = (wave & 3) * 32;    // 0..96
    const int hi = lane >> 4;          // 0/1
    const int lr = lane & 15;

    v8f acc[4][2];
#pragma unroll
    for (int i = 0; i < 4; ++i)
#pragma unroll
        for (int j = 0; j < 2; ++j)
#pragma unroll
            for (int r = 0; r < 8; ++r) acc[i][j][r] = 0.0f;

    for (int k0 = 0; k0 < K; k0 += 32) {
        {   // stage 128x32 bf16 tiles of A and B: 16 u16 per thread each
            int row = tid >> 1, seg = (tid & 1) * 16;
            const v4u* ga = reinterpret_cast<const v4u*>(&A[(size_t)(m0 + row) * K + k0 + seg]);
            v4u* la = reinterpret_cast<v4u*>(&As[row][seg]);
            la[0] = ga[0]; la[1] = ga[1];
            const v4u* gb = reinterpret_cast<const v4u*>(&Bm[(size_t)(n0 + row) * K + k0 + seg]);
            v4u* lb = reinterpret_cast<v4u*>(&Bs[row][seg]);
            lb[0] = gb[0]; lb[1] = gb[1];
        }
        __syncthreads();

        Frag af[4], bfr[2];
#pragma unroll
        for (int i = 0; i < 4; ++i) {
            const u16* p = &As[wm + i * 16 + lr][0];
            af[i].q[0] = *reinterpret_cast<const v4u*>(p + hi * 16);
            af[i].q[1] = *reinterpret_cast<const v4u*>(p + hi * 16 + 8);
        }
#pragma unroll
        for (int j = 0; j < 2; ++j) {
            const u16* p = &Bs[wn + j * 16 + lr][hi * 16];
            bfr[j].q[0] = reinterpret_cast<const v4u*>(p)[0];
            bfr[j].q[1] = reinterpret_cast<const v4u*>(p)[1];
        }
#pragma unroll
        for (int i = 0; i < 4; ++i)
#pragma unroll
            for (int j = 0; j < 2; ++j)
                acc[i][j] = __builtin_amdgcn_wmma_f32_16x16x32_bf16(
                    false, af[i].v, false, bfr[j].v, (short)0, acc[i][j], false, false);
        __syncthreads();
    }

#pragma unroll
    for (int i = 0; i < 4; ++i)
#pragma unroll
        for (int j = 0; j < 2; ++j) {
            int n = n0 + wn + j * 16 + lr;
            int mb = m0 + wm + i * 16 + hi * 8;
            float* cp = C + (size_t)mb * N + n;
#pragma unroll
            for (int r = 0; r < 8; ++r) cp[(size_t)r * N] = acc[i][j][r];
        }
}

// =====================================================================
// xpos rotary + reshape to [b,h,t,hd] bf16 for q,k,v
// one block per (b,t) row of qkv[B*T,4096]
// =====================================================================
__global__ __launch_bounds__(256) void pack_qkv_kernel(
    const float* __restrict__ qkv,
    u16* __restrict__ qb, u16* __restrict__ kb, u16* __restrict__ vb)
{
    const int bt = blockIdx.x;
    const int b = bt >> 11, t = bt & 2047;
    const float* row = qkv + (size_t)bt * 4096;
    const float tf = (float)t;
    const float LOG2_1E4 = 13.287712379549449f; // log2(10000)

    for (int i = threadIdx.x; i < 512; i += 256) {
        float sv   = (2.0f * i + 409.6f) / 1433.6f;          // scale_vec
        float sc   = exp2f((tf / 512.0f) * log2f(sv));       // sv^(t/512)
        float invf = exp2f(-((float)i / 512.0f) * LOG2_1E4);
        float s, c;
        sincosf(tf * invf, &s, &c);
        float q0 = row[2 * i], q1 = row[2 * i + 1];
        float k0 = row[1024 + 2 * i], k1 = row[1024 + 2 * i + 1];
        float qa = q0 * c * sc - q1 * s * sc;
        float qc2 = q1 * c * sc + q0 * s * sc;
        float isc = 1.0f / sc;
        float ka = k0 * c * isc - k1 * s * isc;
        float kc2 = k1 * c * isc + k0 * s * isc;
        int j0 = 2 * i, h = j0 >> 7, hd = j0 & 127;
        size_t base = ((size_t)(b * NH + h) * Tlen + t) * HD_QK;
        qb[base + hd] = f2bf(qa); qb[base + hd + 1] = f2bf(qc2);
        kb[base + hd] = f2bf(ka); kb[base + hd + 1] = f2bf(kc2);
    }
    for (int j = threadIdx.x; j < 2048; j += 256) {
        int h = j >> 8, hd = j & 255;
        vb[((size_t)(b * NH + h) * Tlen + t) * HD_V + hd] = f2bf(row[2048 + j]);
    }
}

// =====================================================================
// Retention: flash-style, one block per (query-chunk of 64, h, b)
// 128 thr = 4 waves, each wave owns 16 query rows, full HD_V=256 output
// =====================================================================
__global__ __launch_bounds__(128) void retention_kernel(
    const u16* __restrict__ qb, const u16* __restrict__ kb, const u16* __restrict__ vb,
    const float* __restrict__ decay, float* __restrict__ X)
{
    const int qc = blockIdx.x, h = blockIdx.y, b = blockIdx.z;
    const int bh = b * NH + h;
    const int tid = threadIdx.x, wave = tid >> 5, lane = tid & 31;
    const int hi = lane >> 4, lr = lane & 15;
    const float ld2 = log2f(decay[h]);
    const float qscale = 0.08838834764831845f;   // 1/sqrt(128)

    __shared__ __align__(16) u16 Qs[64][136];
    __shared__ __align__(16) u16 Ks[64][136];
    __shared__ __align__(16) u16 Vt[256][72];   // v transposed: [vd][s]
    __shared__ __align__(16) u16 Ss[64][72];    // decayed scores, bf16

    // stage Q chunk once
    const u16* qrow = qb + ((size_t)bh * Tlen + qc * 64) * HD_QK;
    for (int it = tid; it < 1024; it += 128) {         // 64 rows x 16 v4u
        int r = it >> 4, c = (it & 15) * 8;
        *reinterpret_cast<v4u*>(&Qs[r][c]) =
            *reinterpret_cast<const v4u*>(&qrow[(size_t)r * HD_QK + c]);
    }
    __syncthreads();

    v8f o[16];
#pragma unroll
    for (int n = 0; n < 16; ++n)
#pragma unroll
        for (int r = 0; r < 8; ++r) o[n][r] = 0.0f;

    for (int sc = 0; sc <= qc; ++sc) {
        // stage K chunk and transposed V chunk
        const u16* krow = kb + ((size_t)bh * Tlen + sc * 64) * HD_QK;
        for (int it = tid; it < 1024; it += 128) {
            int r = it >> 4, c = (it & 15) * 8;
            *reinterpret_cast<v4u*>(&Ks[r][c]) =
                *reinterpret_cast<const v4u*>(&krow[(size_t)r * HD_QK + c]);
        }
        const u16* vrow = vb + ((size_t)bh * Tlen + sc * 64) * HD_V;
        for (int it = tid; it < 8192; it += 128) {      // 64 s x 128 dword-pairs
            int s = it >> 7, vd = (it & 127) * 2;
            unsigned u = *reinterpret_cast<const unsigned*>(&vrow[(size_t)s * HD_V + vd]);
            Vt[vd][s]     = (u16)(u & 0xffffu);
            Vt[vd + 1][s] = (u16)(u >> 16);
        }
        __syncthreads();

        // ---- S = Q K^T (16x64 per wave), K dim = 128 in 4 steps ----
        v8f sacc[4];
#pragma unroll
        for (int j = 0; j < 4; ++j)
#pragma unroll
            for (int r = 0; r < 8; ++r) sacc[j][r] = 0.0f;

#pragma unroll
        for (int kk = 0; kk < 4; ++kk) {
            Frag aq;
            const u16* pa = &Qs[wave * 16 + lr][kk * 32];
            aq.q[0] = *reinterpret_cast<const v4u*>(pa + hi * 16);
            aq.q[1] = *reinterpret_cast<const v4u*>(pa + hi * 16 + 8);
#pragma unroll
            for (int j = 0; j < 4; ++j) {
                Frag bk;
                const u16* pb = &Ks[j * 16 + lr][kk * 32 + hi * 16];
                bk.q[0] = reinterpret_cast<const v4u*>(pb)[0];
                bk.q[1] = reinterpret_cast<const v4u*>(pb)[1];
                sacc[j] = __builtin_amdgcn_wmma_f32_16x16x32_bf16(
                    false, aq.v, false, bk.v, (short)0, sacc[j], false, false);
            }
        }

        // ---- decay + causal mask on C-fragments, write bf16 S to LDS ----
#pragma unroll
        for (int j = 0; j < 4; ++j) {
            int sg = sc * 64 + j * 16 + lr;
#pragma unroll
            for (int r = 0; r < 8; ++r) {
                int m  = r + hi * 8;
                int tg = qc * 64 + wave * 16 + m;
                float f = (sg <= tg) ? qscale * exp2f(ld2 * (float)(tg - sg)) : 0.0f;
                Ss[wave * 16 + m][j * 16 + lr] = f2bf(sacc[j][r] * f);
            }
        }

        // ---- out += S(16x64) @ V(64x256): B operand is Vt[n][k] ----
#pragma unroll
        for (int kk = 0; kk < 2; ++kk) {
            Frag as;
            const u16* pa = &Ss[wave * 16 + lr][kk * 32];
            as.q[0] = *reinterpret_cast<const v4u*>(pa + hi * 16);
            as.q[1] = *reinterpret_cast<const v4u*>(pa + hi * 16 + 8);
#pragma unroll
            for (int n = 0; n < 16; ++n) {
                Frag bv;
                const u16* pb = &Vt[n * 16 + lr][kk * 32 + hi * 16];
                bv.q[0] = reinterpret_cast<const v4u*>(pb)[0];
                bv.q[1] = reinterpret_cast<const v4u*>(pb)[1];
                o[n] = __builtin_amdgcn_wmma_f32_16x16x32_bf16(
                    false, as.v, false, bv.v, (short)0, o[n], false, false);
            }
        }
        __syncthreads();   // protect Ks/Vt for next chunk
    }

    // store X[b,h,t,vd] f32
#pragma unroll
    for (int n = 0; n < 16; ++n) {
        int vd = n * 16 + lr;
#pragma unroll
        for (int r = 0; r < 8; ++r) {
            int t = qc * 64 + wave * 16 + r + hi * 8;
            X[((size_t)bh * Tlen + t) * HD_V + vd] = o[n][r];
        }
    }
}

// =====================================================================
// curr_kv[b,h,d,vd] = sum_s decay^(T-1-s) k[s,d] v[s,vd]
// one block (256 thr, 8 waves, wave tile 64x64) per (b,h)
// =====================================================================
__global__ __launch_bounds__(256) void currkv_kernel(
    const u16* __restrict__ kb, const u16* __restrict__ vb,
    const float* __restrict__ decay, float* __restrict__ out)
{
    const int bh = blockIdx.x, h = bh & 7;
    const int tid = threadIdx.x, wave = tid >> 5, lane = tid & 31;
    const int hi = lane >> 4, lr = lane & 15;
    const int wm = (wave >> 2) * 64;   // d tile base
    const int wn = (wave & 3) * 64;    // vd tile base
    const float ld2 = log2f(decay[h]);

    __shared__ __align__(16) u16 kT[128][40];   // [d][s] decay-weighted
    __shared__ __align__(16) u16 vT[256][40];   // [vd][s]

    v8f acc[4][4];
#pragma unroll
    for (int i = 0; i < 4; ++i)
#pragma unroll
        for (int j = 0; j < 4; ++j)
#pragma unroll
            for (int r = 0; r < 8; ++r) acc[i][j][r] = 0.0f;

    for (int s0 = 0; s0 < Tlen; s0 += 32) {
        for (int it = tid; it < 4096; it += 256) {      // 32 s x 128 d
            int s = it >> 7, d = it & 127;
            float w = exp2f(ld2 * (float)(Tlen - 1 - (s0 + s)));
            float v = bf2f(kb[((size_t)bh * Tlen + s0 + s) * HD_QK + d]);
            kT[d][s] = f2bf(v * w);
        }
        for (int it = tid; it < 8192; it += 256) {      // 32 s x 256 vd
            int s = it >> 8, vd = it & 255;
            vT[vd][s] = vb[((size_t)bh * Tlen + s0 + s) * HD_V + vd];
        }
        __syncthreads();

        Frag af[4], bfr[4];
#pragma unroll
        for (int i = 0; i < 4; ++i) {
            const u16* p = &kT[wm + i * 16 + lr][0];
            af[i].q[0] = *reinterpret_cast<const v4u*>(p + hi * 16);
            af[i].q[1] = *reinterpret_cast<const v4u*>(p + hi * 16 + 8);
        }
#pragma unroll
        for (int j = 0; j < 4; ++j) {
            const u16* p = &vT[wn + j * 16 + lr][hi * 16];
            bfr[j].q[0] = reinterpret_cast<const v4u*>(p)[0];
            bfr[j].q[1] = reinterpret_cast<const v4u*>(p)[1];
        }
#pragma unroll
        for (int i = 0; i < 4; ++i)
#pragma unroll
            for (int j = 0; j < 4; ++j)
                acc[i][j] = __builtin_amdgcn_wmma_f32_16x16x32_bf16(
                    false, af[i].v, false, bfr[j].v, (short)0, acc[i][j], false, false);
        __syncthreads();
    }

#pragma unroll
    for (int i = 0; i < 4; ++i)
#pragma unroll
        for (int j = 0; j < 4; ++j) {
            int vd = wn + j * 16 + lr;
#pragma unroll
            for (int r = 0; r < 8; ++r) {
                int d = wm + i * 16 + r + hi * 8;
                out[(size_t)bh * (HD_QK * HD_V) + (size_t)d * HD_V + vd] = acc[i][j][r];
            }
        }
}

// =====================================================================
// groupnorm over HD_V per (b,t,h) + silu gate; output bf16 rows for final GEMM
// one wave per group, 8 groups per block
// =====================================================================
__global__ __launch_bounds__(256) void gn_gate_kernel(
    const float* __restrict__ X, const float* __restrict__ gate_pre, u16* __restrict__ gb)
{
    int g = blockIdx.x * 8 + (threadIdx.x >> 5);
    int lane = threadIdx.x & 31;
    int bt = g >> 3, h = g & 7;
    int b = bt >> 11, t = bt & 2047;
    const float* xr = X + ((size_t)(b * NH + h) * Tlen + t) * HD_V;

    float x[8], sum = 0.0f, sq = 0.0f;
#pragma unroll
    for (int i = 0; i < 8; ++i) {
        x[i] = xr[lane + i * 32];
        sum += x[i]; sq += x[i] * x[i];
    }
#pragma unroll
    for (int off = 16; off > 0; off >>= 1) {
        sum += __shfl_xor(sum, off, 32);
        sq  += __shfl_xor(sq,  off, 32);
    }
    float mu = sum * (1.0f / HD_V);
    float var = sq * (1.0f / HD_V) - mu * mu;
    float rstd = rsqrtf(var + 1e-5f);

    size_t base = (size_t)bt * VD + h * HD_V;
#pragma unroll
    for (int i = 0; i < 8; ++i) {
        int vd = lane + i * 32;
        float gt = gate_pre[base + vd];
        float sil = gt / (1.0f + expf(-gt));
        gb[base + vd] = f2bf((x[i] - mu) * rstd * sil);
    }
}

// =====================================================================
// host launcher
// =====================================================================
extern "C" void kernel_launch(void* const* d_in, const int* in_sizes, int n_in,
                              void* d_out, int out_size, void* d_ws, size_t ws_size,
                              hipStream_t stream) {
    (void)in_sizes; (void)n_in; (void)out_size; (void)ws_size;
    const float* hs      = (const float*)d_in[0];
    const float* w_qkv   = (const float*)d_in[1];
    const float* w_gated = (const float*)d_in[2];
    const float* w_proj  = (const float*)d_in[3];
    const float* decay   = (const float*)d_in[4];

    float* y   = (float*)d_out;                       // [B,T,HS]
    float* ckv = y + (size_t)Bsz * Tlen * HS;         // [B,NH,128,256]

    char* ws = (char*)d_ws;
    const size_t MB = 1u << 20;
    u16*   hsb   = (u16*)(ws + 0);         //  8 MB  hidden bf16 [4096,1024]
    u16*   wqkvb = (u16*)(ws + 8  * MB);   //  8 MB
    u16*   wgb   = (u16*)(ws + 16 * MB);   //  4 MB
    u16*   wpb   = (u16*)(ws + 20 * MB);   //  4 MB
    u16*   qb    = (u16*)(ws + 24 * MB);   //  8 MB  [b,h,t,128] bf16
    u16*   kb    = (u16*)(ws + 32 * MB);   //  8 MB
    u16*   vb    = (u16*)(ws + 40 * MB);   // 16 MB  [b,h,t,256] bf16
    float* gatep = (float*)(ws + 56 * MB); // 32 MB  [4096,2048] f32
    float* qkv   = (float*)(ws + 88 * MB); // 64 MB  [4096,4096] f32
    float* X     = qkv;                    // alias: retention out reuses qkv
    u16*   gb    = qb;                     // alias: gated/normed rows reuse q/k (after currkv)

    // 1) bf16 packs
    f32_to_bf16_kernel<<<2048, 256, 0, stream>>>(hs,      hsb,   Bsz * Tlen * HS);
    f32_to_bf16_kernel<<<2048, 256, 0, stream>>>(w_qkv,   wqkvb, (2 * QK + VD) * HS);
    f32_to_bf16_kernel<<<1024, 256, 0, stream>>>(w_gated, wgb,   VD * HS);
    f32_to_bf16_kernel<<<1024, 256, 0, stream>>>(w_proj,  wpb,   HS * VD);

    // 2) qkv = hs @ w_qkv^T   (M=4096, N=4096, K=1024)
    gemm_bf16_kernel<<<dim3(32, 32), 256, 0, stream>>>(hsb, wqkvb, qkv, 4096, 4096, 1024);
    // 3) gate_pre = hs @ w_gated^T (M=4096, N=2048, K=1024)
    gemm_bf16_kernel<<<dim3(16, 32), 256, 0, stream>>>(hsb, wgb, gatep, 4096, 2048, 1024);

    // 4) xpos + head pack
    pack_qkv_kernel<<<Bsz * Tlen, 256, 0, stream>>>(qkv, qb, kb, vb);

    // 5) retention (writes X over qkv)
    retention_kernel<<<dim3(Tlen / 64, NH, Bsz), 128, 0, stream>>>(qb, kb, vb, decay, X);

    // 6) curr_kv (needs kb/vb; must precede gb alias writes)
    currkv_kernel<<<Bsz * NH, 256, 0, stream>>>(kb, vb, decay, ckv);

    // 7) groupnorm + silu gate -> bf16 rows
    gn_gate_kernel<<<(Bsz * Tlen * NH) / 8, 256, 0, stream>>>(X, gatep, gb);

    // 8) y = g @ w_proj^T  (M=4096, N=1024, K=2048)
    gemm_bf16_kernel<<<dim3(8, 32), 256, 0, stream>>>(gb, wpb, y, 4096, 1024, 2048);
}
